// EquiFilter_73899207295097
// MI455X (gfx1250) — matrix operand order repaired and checked by reading, not compiled, not run on hardware
//
#include <hip/hip_runtime.h>

// ---------------------------------------------------------------------------
// EquiFilter for MI455X (gfx1250): bf16 WMMA MLP over 500k edges.
// Layout plan:
//   prep kernel:  d_ws (bf16, transposed N-major/K-contig):
//     B1t  [128 n][128 k]   (Wi1, K padded 112->128)        @ elem 0
//     Brt  [128 n][ 32 k]   (Wr1, K padded 20->32)          @ elem 16384
//     B2t  [128 n][256 k]   ([Wi2;Wr2] stacked on K)        @ elem 20480
//   main kernel:  64 edges / block, 8 waves, wave w owns 16 output cols.
// ---------------------------------------------------------------------------

typedef __attribute__((ext_vector_type(8)))  float  v8f;
typedef __attribute__((ext_vector_type(8)))  __bf16 bf16x8;
typedef __attribute__((ext_vector_type(16))) __bf16 bf16x16;

union Frag {
  bf16x16 f;
  bf16x8  h[2];
};

__device__ __forceinline__ unsigned short f32_to_bf16(float x) {
  unsigned int u = __float_as_uint(x);
  u += 0x7FFFu + ((u >> 16) & 1u);   // round-to-nearest-even
  return (unsigned short)(u >> 16);
}

__device__ __forceinline__ float silu(float x) {
  return x / (1.0f + __expf(-x));
}

// -------------------- weight prep: f32 -> transposed bf16 -------------------
#define OFF_B1 0
#define OFF_BR 16384
#define OFF_B2 20480
#define WS_TOTAL 53248

__global__ void prep_weights(const float* __restrict__ Wi1,
                             const float* __restrict__ Wr1,
                             const float* __restrict__ Wi2,
                             const float* __restrict__ Wr2,
                             unsigned short* __restrict__ ws) {
  int idx = blockIdx.x * 256 + threadIdx.x;
  if (idx >= WS_TOTAL) return;
  float v = 0.0f;
  if (idx < OFF_BR) {                       // B1t[n][k] = Wi1[k][n], k<112
    int n = idx >> 7, k = idx & 127;
    if (k < 112) v = Wi1[k * 128 + n];
  } else if (idx < OFF_B2) {                // Brt[n][k] = Wr1[k][n], k<20
    int j = idx - OFF_BR;
    int n = j >> 5, k = j & 31;
    if (k < 20) v = Wr1[k * 128 + n];
  } else {                                  // B2t[n][k] = [Wi2;Wr2][k][n]
    int j = idx - OFF_B2;
    int n = j >> 8, k = j & 255;
    v = (k < 128) ? Wi2[k * 128 + n] : Wr2[(k - 128) * 128 + n];
  }
  ws[idx] = f32_to_bf16(v);
}

// ------------------------------- main kernel --------------------------------
// LDS row strides (ushorts), padded to de-conflict 64x4B banks.
#define SA 136   // invA: 64 x 128 (stride 136)
#define SR 40    // rbfA: 64 x  32 (stride 40)
#define S2 264   // A2:   64 x 256 (stride 264)

__global__ __launch_bounds__(256) void edge_filter(
    const float* __restrict__ x_equi, const float* __restrict__ rbf,
    const float* __restrict__ bi1, const float* __restrict__ bi2,
    const float* __restrict__ br1, const float* __restrict__ br2,
    const int* __restrict__ cidx, const int* __restrict__ nidx,
    const unsigned short* __restrict__ wsB, float* __restrict__ out,
    int n_edges)
{
  __shared__ alignas(16) unsigned short invA[64 * SA];
  __shared__ alignas(16) unsigned short rbfA[64 * SR];
  __shared__ alignas(16) unsigned short A2[64 * S2];

  const int tid = threadIdx.x;
  const int e0  = blockIdx.x * 64;

  // ---- Phase 1: gather + per-irrep <x_ij, x_ij>, quantize to bf16 in LDS ---
  for (int p = tid; p < 64 * 128; p += 256) {
    int el = p >> 7, k = p & 127;
    unsigned short val = 0;
    if (k < 112) {
      int e = e0 + el; if (e >= n_edges) e = n_edges - 1;
      const float* xi = x_equi + cidx[e] * 240;
      const float* xj = x_equi + nidx[e] * 240;
      int off, d;
      if (k < 64)      { off = k;                 d = 1; }
      else if (k < 96) { off = 64 + (k - 64) * 3; d = 3; }
      else             { off = 160 + (k - 96) * 5; d = 5; }
      float s = 0.0f;
      for (int c = 0; c < d; ++c) {
        float t = xj[off + c] - xi[off + c];
        s += t * t;
      }
      val = f32_to_bf16(s);
    }
    invA[el * SA + k] = val;
  }
  for (int p = tid; p < 64 * 32; p += 256) {
    int el = p >> 5, k = p & 31;
    unsigned short val = 0;
    if (k < 20) {
      int e = e0 + el; if (e >= n_edges) e = n_edges - 1;
      val = f32_to_bf16(rbf[e * 20 + k]);
    }
    rbfA[el * SR + k] = val;
  }
  __syncthreads();

  // Fragment geometry (wave32):
  //  A (16x32 bf16):  m = lane%16;  k-halves at krow and krow+16, krow = lane<16?0:8
  //  B (32x16 bf16):  n = lane%16;  k-halves at kbB and kbB+16, kbB = lane<16?0:8
  //  C/D (16x16 f32): n = lane%16;  m = vgpr + (lane<16?0:8)
  const int lane  = tid & 31;
  const int wv    = tid >> 5;
  const int n0    = wv * 16;
  const int nlane = n0 + (lane & 15);
  const int mlane = lane & 15;
  const int krow  = (lane < 16) ? 0 : 8;
  const int kbB   = (lane < 16) ? 0 : 8;
  const int mhi   = (lane < 16) ? 0 : 8;

  // ---- Phase 2: h1 = silu(inv@Wi1+bi1), r1 = silu(rbf@Wr1+br1) -> A2 ------
  Frag b1[4];
#pragma unroll
  for (int ks = 0; ks < 4; ++ks) {
    const unsigned short* bp = wsB + OFF_B1 + nlane * 128 + ks * 32 + kbB;
    b1[ks].h[0] = *(const bf16x8*)(bp);
    b1[ks].h[1] = *(const bf16x8*)(bp + 16);
  }
  Frag brf;
  {
    const unsigned short* bp = wsB + OFF_BR + nlane * 32 + kbB;
    brf.h[0] = *(const bf16x8*)(bp);
    brf.h[1] = *(const bf16x8*)(bp + 16);
  }
  const float bias1 = bi1[nlane];
  const float biasr = br1[nlane];

#pragma unroll
  for (int mt = 0; mt < 4; ++mt) {
    const int m0 = mt * 16;
    const int mA = m0 + mlane;
    v8f acc  = {0.f, 0.f, 0.f, 0.f, 0.f, 0.f, 0.f, 0.f};
    v8f accr = {0.f, 0.f, 0.f, 0.f, 0.f, 0.f, 0.f, 0.f};
#pragma unroll
    for (int ks = 0; ks < 4; ++ks) {
      Frag a;
      const unsigned short* ap = invA + mA * SA + ks * 32 + krow;
      a.h[0] = *(const bf16x8*)(ap);
      a.h[1] = *(const bf16x8*)(ap + 16);
      acc = __builtin_amdgcn_wmma_f32_16x16x32_bf16(
          false, a.f, false, b1[ks].f, (short)0, acc, false, false);
    }
    {
      Frag a;
      const unsigned short* ap = rbfA + mA * SR + krow;
      a.h[0] = *(const bf16x8*)(ap);
      a.h[1] = *(const bf16x8*)(ap + 16);
      accr = __builtin_amdgcn_wmma_f32_16x16x32_bf16(
          false, a.f, false, brf.f, (short)0, accr, false, false);
    }
#pragma unroll
    for (int r = 0; r < 8; ++r) {
      const int m = m0 + r + mhi;
      A2[m * S2 + nlane]       = f32_to_bf16(silu(acc[r] + bias1));
      A2[m * S2 + 128 + nlane] = f32_to_bf16(silu(accr[r] + biasr));
    }
  }
  __syncthreads();

  // ---- Phase 3: out = [h1|r1] @ [Wi2;Wr2] + (bi2+br2), K = 256 -------------
  Frag b2[8];
#pragma unroll
  for (int ks = 0; ks < 8; ++ks) {
    const unsigned short* bp = wsB + OFF_B2 + nlane * 256 + ks * 32 + kbB;
    b2[ks].h[0] = *(const bf16x8*)(bp);
    b2[ks].h[1] = *(const bf16x8*)(bp + 16);
  }
  const float bias2 = bi2[nlane] + br2[nlane];

#pragma unroll
  for (int mt = 0; mt < 4; ++mt) {
    const int m0 = mt * 16;
    const int mA = m0 + mlane;
    v8f acc = {0.f, 0.f, 0.f, 0.f, 0.f, 0.f, 0.f, 0.f};
#pragma unroll
    for (int ks = 0; ks < 8; ++ks) {
      Frag a;
      const unsigned short* ap = A2 + mA * S2 + ks * 32 + krow;
      a.h[0] = *(const bf16x8*)(ap);
      a.h[1] = *(const bf16x8*)(ap + 16);
      acc = __builtin_amdgcn_wmma_f32_16x16x32_bf16(
          false, a.f, false, b2[ks].f, (short)0, acc, false, false);
    }
#pragma unroll
    for (int r = 0; r < 8; ++r) {
      const int m = m0 + r + mhi;
      const int e = e0 + m;
      if (e < n_edges) out[e * 128 + nlane] = acc[r] + bias2;
    }
  }
}

// ------------------------------- launcher -----------------------------------
extern "C" void kernel_launch(void* const* d_in, const int* in_sizes, int n_in,
                              void* d_out, int out_size, void* d_ws, size_t ws_size,
                              hipStream_t stream) {
  const float* x_equi = (const float*)d_in[0];
  const float* rbf    = (const float*)d_in[1];
  const float* Wi1    = (const float*)d_in[2];
  const float* bi1    = (const float*)d_in[3];
  const float* Wi2    = (const float*)d_in[4];
  const float* bi2    = (const float*)d_in[5];
  const float* Wr1    = (const float*)d_in[6];
  const float* br1    = (const float*)d_in[7];
  const float* Wr2    = (const float*)d_in[8];
  const float* br2    = (const float*)d_in[9];
  const int*   cidx   = (const int*)d_in[10];
  const int*   nidx   = (const int*)d_in[11];
  unsigned short* wsB = (unsigned short*)d_ws;
  float* out = (float*)d_out;
  const int n_edges = in_sizes[10];

  prep_weights<<<(WS_TOTAL + 255) / 256, 256, 0, stream>>>(Wi1, Wr1, Wi2, Wr2, wsB);

  const int nblocks = (n_edges + 63) / 64;
  edge_filter<<<nblocks, 256, 0, stream>>>(x_equi, rbf, bi1, bi2, br1, br2,
                                           cidx, nidx, wsB, out, n_edges);
}
